// LSTMBlock_56934086476261
// MI455X (gfx1250) — matrix operand order repaired and checked by reading, not compiled, not run on hardware
//
#include <hip/hip_runtime.h>

// LSTM for MI455X (gfx1250): bf16 WMMA GEMMs + fp32 accumulate.
// Phase 1: cast/pack operands; x-projection GEMM with LDS-staged B tile
//          (async global->LDS DMA) and software-pipelined A loads.
// Phase 2: persistent-scan kernel, 64 WGs, LDS-resident recurrent weights
//          (async-DMA staged), register-resident cell state, device-wide
//          barrier per timestep.

#define B_ 32
#define T_ 512
#define I_ 1024
#define H_ 1024
#define G_ 4096   // 4 gates * H
#define NWG_ 64   // persistent workgroups in scan

typedef __attribute__((ext_vector_type(16))) __bf16 v16bf;
typedef __attribute__((ext_vector_type(8)))  float  v8f;

union Frag16 { v16bf v; uint4 q[2]; };

static __device__ __forceinline__ unsigned short f2bf(float f) {
  unsigned int u = __float_as_uint(f);
  u += 0x7FFFu + ((u >> 16) & 1u);       // round-to-nearest-even
  return (unsigned short)(u >> 16);
}
static __device__ __forceinline__ float sigmoidf_(float x) {
  return 1.0f / (1.0f + __expf(-x));
}

// gfx1250 async global->LDS copy of 16 bytes per lane (tracked by ASYNCcnt).
// dst_lds = byte offset within the workgroup's LDS allocation.
static __device__ __forceinline__ void async_ld16(unsigned dst_lds, const void* src) {
  asm volatile("global_load_async_to_lds_b128 %0, %1, off"
               :: "v"(dst_lds), "v"((unsigned long long)(size_t)src)
               : "memory");
}
static __device__ __forceinline__ void async_wait0() {
  asm volatile("s_wait_asynccnt 0" ::: "memory");
}

// ---------------- packing / casting kernels ----------------

__global__ void cast_x_kernel(const float* __restrict__ x,
                              unsigned short* __restrict__ xbf, int n) {
  int i = blockIdx.x * blockDim.x + threadIdx.x;
  if (i < n) xbf[i] = f2bf(x[i]);
}

struct WPtrs { const float* p[8]; };
__global__ void pack_w_kernel(WPtrs wp, unsigned short* __restrict__ wx,
                              unsigned short* __restrict__ wh) {
  int i = blockIdx.x * blockDim.x + threadIdx.x;   // 0 .. 8M-1
  int m = i >> 20;
  int off = i & ((1 << 20) - 1);
  unsigned short v = f2bf(wp.p[m][off]);
  if (m < 4) wx[(size_t)m * (1 << 20) + off] = v;          // W_x{i,f,c,o}
  else       wh[(size_t)(m - 4) * (1 << 20) + off] = v;    // W_h{i,f,c,o}
}

struct BPtrs { const float* p[4]; };
__global__ void pack_b_kernel(BPtrs bp, float* __restrict__ bias) {
  int i = blockIdx.x * blockDim.x + threadIdx.x;
  if (i < G_) bias[i] = bp.p[i >> 10][i & (H_ - 1)];
}

__global__ void init_kernel(unsigned short* __restrict__ hbuf,
                            unsigned int* __restrict__ bar) {
  for (int i = threadIdx.x; i < 2 * B_ * H_; i += blockDim.x) hbuf[i] = 0;
  if (threadIdx.x < 32) bar[threadIdx.x] = 0;
}

// ---------------- phase 1: x-projection GEMM ----------------
// P[t][b][g] = bias[g] + sum_k x[b][t][k] * W_x[g][k]   (fp32 out, bf16 WMMA)
// Block = 8 waves = 8 M-subtiles (16 rows each) x one 64-col N-tile.
// B slice (64 rows x 1024 K = 128 KB) staged in LDS once via async DMA,
// shared by all 8 waves; A fragments pipelined one k-step ahead.

__global__ void __launch_bounds__(256) xproj_kernel(
    const unsigned short* __restrict__ xbf,   // [B*T][I] bf16
    const unsigned short* __restrict__ wx,    // [G][I]   bf16
    const float* __restrict__ bias,           // [G]
    float* __restrict__ P)                    // [T][B][G]
{
  extern __shared__ char smem[];
  unsigned short* b_lds = (unsigned short*)smem;   // [64][1024] bf16 = 128 KB

  const int lane = threadIdx.x & 31;
  const int wave = threadIdx.x >> 5;
  const int blkM = blockIdx.x >> 6;     // 128 block-rows (8 M-tiles each)
  const int n0   = (blockIdx.x & 63) << 6;
  const int hl   = lane >> 4;           // lane half
  const int ln   = lane & 15;

  // Stage B slice into LDS: 8192 x 16B chunks, 32 per thread, async DMA.
  {
#pragma unroll 8
    for (int it = 0; it < 32; ++it) {
      const int idx = it * 256 + threadIdx.x;   // 16B-unit index
      const int row = idx >> 7;                 // 128 units per 2KB row
      const int u   = idx & 127;
      async_ld16((unsigned)(row * 2048 + u * 16),
                 wx + (size_t)(n0 + row) * I_ + u * 8);
    }
    async_wait0();
  }
  __syncthreads();

  const int R  = (blkM * 8 + wave) << 4;  // M base row
  const int b  = R >> 9;                  // T=512 rows per batch entry
  const int t0 = R & 511;
  const unsigned short* aRow = xbf + (size_t)(R + ln) * I_;

  v8f acc[4];
#pragma unroll
  for (int nb = 0; nb < 4; ++nb) {
    float bv = bias[n0 + nb * 16 + ln];
#pragma unroll
    for (int r = 0; r < 8; ++r) acc[nb][r] = bv;
  }

  // A 16x32 bf16: lanes 0-15 hold K{0..7,16..23}, lanes 16-31 K{8..15,24..31}
  Frag16 a_cur, a_nxt;
  a_cur.q[0] = *(const uint4*)(aRow + hl * 8);
  a_cur.q[1] = *(const uint4*)(aRow + hl * 8 + 16);

#pragma unroll 4
  for (int kt = 0; kt < I_ / 32; ++kt) {
    if (kt < I_ / 32 - 1) {                 // prefetch next A fragment
      const int ka = (kt + 1) * 32 + hl * 8;
      a_nxt.q[0] = *(const uint4*)(aRow + ka);
      a_nxt.q[1] = *(const uint4*)(aRow + ka + 16);
    }
    const int kb = kt * 32 + hl * 8;        // B 32x16: same K order as A
#pragma unroll
    for (int nb = 0; nb < 4; ++nb) {
      Frag16 bq;
      const unsigned short* br = b_lds + (nb * 16 + ln) * I_ + kb;
      bq.q[0] = *(const uint4*)br;
      bq.q[1] = *(const uint4*)(br + 16);
      acc[nb] = __builtin_amdgcn_wmma_f32_16x16x32_bf16(
          false, a_cur.v, false, bq.v, (short)0, acc[nb], false, false);
    }
    a_cur = a_nxt;
  }

#pragma unroll
  for (int nb = 0; nb < 4; ++nb) {
    const int n = n0 + nb * 16 + ln;
#pragma unroll
    for (int r = 0; r < 8; ++r) {
      const int m = r + hl * 8;             // C/D: reg r -> M = r (+8 hi half)
      P[((size_t)(t0 + m) * B_ + b) * G_ + n] = acc[nb][r];
    }
  }
}

// ---------------- phase 2: persistent recurrent scan ----------------
// 64 WGs x 256 threads. WG owns 16 H-columns (all 4 gates = 64 gate rows).
// Weights (128 KB bf16) async-DMA'd to LDS once; h ping-pongs in global.

__global__ void __launch_bounds__(256) lstm_scan_kernel(
    const float* __restrict__ P,              // [T][B][G]
    const unsigned short* __restrict__ wh,    // [G][H] bf16
    unsigned short* __restrict__ hbuf,        // 2 x [B][H] bf16 ping-pong
    float* __restrict__ out,                  // [B][T][H] ++ [B][H]
    unsigned int* __restrict__ bar)
{
  extern __shared__ char smem[];
  unsigned short* w_lds = (unsigned short*)smem;                     // [64][1024]
  float* gates = (float*)(smem + 64 * H_ * sizeof(unsigned short));  // [64][32]

  const int wg = blockIdx.x;
  const int col_base = wg << 4;
  const int tid = threadIdx.x;
  const int lane = tid & 31;
  const int wave = tid >> 5;
  const int mt = wave >> 2;       // batch tile (0..1)
  const int nt = wave & 3;        // gate (i,f,c~,o)
  const int hl = lane >> 4;
  const int ln = lane & 15;

  // Stage recurrent-weight slice into LDS once via async DMA.
  {
    const int row  = tid >> 2;            // 0..63 (gate*16+col)
    const int part = (tid & 3) * 256;     // ushort offset within row
    const int grow = (row >> 4) * H_ + col_base + (row & 15);
    const unsigned short* src = wh + (size_t)grow * H_ + part;
    const unsigned dstb = (unsigned)(row * 2048 + part * 2);
#pragma unroll 8
    for (int i = 0; i < 32; ++i)
      async_ld16(dstb + i * 16, src + i * 8);
    async_wait0();
  }
  __syncthreads();

  float creg[2] = {0.f, 0.f};     // register-resident cell state (2 elems/thread)
  const int e0 = tid << 1;

  for (int t = 0; t < T_; ++t) {
    const unsigned short* h_rd = hbuf + (size_t)(t & 1) * (B_ * H_);
    unsigned short* h_wr = hbuf + (size_t)((t + 1) & 1) * (B_ * H_);

    // C init = precomputed x-projection (+bias already folded in).
    const int ngl = nt * H_ + col_base + ln;
    v8f acc0, acc1;
#pragma unroll
    for (int r = 0; r < 8; ++r) {
      const int m = (mt << 4) + r + hl * 8;
      acc0[r] = P[((size_t)t * B_ + m) * G_ + ngl];
      acc1[r] = 0.f;
    }
    if (lane == 0 && t + 1 < T_)
      __builtin_prefetch(&P[((size_t)(t + 1) * B_ + (mt << 4)) * G_ + nt * H_ + col_base], 0, 1);

    const unsigned short* aRow = h_rd + (size_t)((mt << 4) + ln) * H_;
    const unsigned short* bRow = w_lds + (nt * 16 + ln) * H_;

    // Two accumulator chains (break WMMA RAW), A-fragment pipelined 1 ahead.
    Frag16 aC, aN;
    aC.q[0] = *(const uint4*)(aRow + hl * 8);
    aC.q[1] = *(const uint4*)(aRow + hl * 8 + 16);
#pragma unroll 2
    for (int kt = 0; kt < H_ / 32; ++kt) {
      if (kt < H_ / 32 - 1) {
        const int ka = (kt + 1) * 32 + hl * 8;
        aN.q[0] = *(const uint4*)(aRow + ka);
        aN.q[1] = *(const uint4*)(aRow + ka + 16);
      }
      Frag16 bq;
      const int kb = kt * 32 + hl * 8;
      bq.q[0] = *(const uint4*)(bRow + kb);
      bq.q[1] = *(const uint4*)(bRow + kb + 16);
      if (kt & 1)
        acc1 = __builtin_amdgcn_wmma_f32_16x16x32_bf16(
            false, aC.v, false, bq.v, (short)0, acc1, false, false);
      else
        acc0 = __builtin_amdgcn_wmma_f32_16x16x32_bf16(
            false, aC.v, false, bq.v, (short)0, acc0, false, false);
      aC = aN;
    }

    // Activation + publish to LDS gate buffer [gate*16+col][b].
    {
      float* grow = gates + (nt * 16 + ln) * 32;
#pragma unroll
      for (int r = 0; r < 8; ++r) {
        const int m = (mt << 4) + r + hl * 8;
        float v = acc0[r] + acc1[r];
        v = (nt == 2) ? tanhf(v) : sigmoidf_(v);
        grow[m] = v;
      }
    }
    __syncthreads();

    // Cell update: each thread owns 2 fixed (col,b) elements across the scan.
#pragma unroll
    for (int j = 0; j < 2; ++j) {
      const int e = e0 + j;
      const int col = e >> 5;
      const int b = e & 31;
      const float iv = gates[(col)      * 32 + b];
      const float fv = gates[(16 + col) * 32 + b];
      const float gv = gates[(32 + col) * 32 + b];
      const float ov = gates[(48 + col) * 32 + b];
      const float c = fv * creg[j] + iv * gv;
      creg[j] = c;
      const float h = ov * tanhf(c);
      out[((size_t)b * T_ + t) * H_ + col_base + col] = h;
      h_wr[(size_t)b * H_ + col_base + col] = f2bf(h);
      if (t == T_ - 1)
        out[(size_t)B_ * T_ * H_ + (size_t)b * H_ + col_base + col] = h;
    }

    // Device-wide barrier: h_{t+1} visible before any WG starts step t+1.
    __threadfence();
    __syncthreads();
    if (tid == 0) {
      unsigned int g = __hip_atomic_load(&bar[1], __ATOMIC_ACQUIRE, __HIP_MEMORY_SCOPE_AGENT);
      if (atomicAdd(&bar[0], 1u) == NWG_ - 1u) {
        __hip_atomic_store(&bar[0], 0u, __ATOMIC_RELAXED, __HIP_MEMORY_SCOPE_AGENT);
        __hip_atomic_store(&bar[1], g + 1u, __ATOMIC_RELEASE, __HIP_MEMORY_SCOPE_AGENT);
      } else {
        while (__hip_atomic_load(&bar[1], __ATOMIC_ACQUIRE, __HIP_MEMORY_SCOPE_AGENT) == g)
          __builtin_amdgcn_s_sleep(1);
      }
    }
    __syncthreads();
  }
}

// ---------------- launch ----------------

extern "C" void kernel_launch(void* const* d_in, const int* in_sizes, int n_in,
                              void* d_out, int out_size, void* d_ws, size_t ws_size,
                              hipStream_t stream) {
  const float* x   = (const float*)d_in[0];
  const float* Wxi = (const float*)d_in[1];
  const float* Whi = (const float*)d_in[2];
  const float* bi  = (const float*)d_in[3];
  const float* Wxf = (const float*)d_in[4];
  const float* Whf = (const float*)d_in[5];
  const float* bf_ = (const float*)d_in[6];
  const float* Wxc = (const float*)d_in[7];
  const float* Whc = (const float*)d_in[8];
  const float* bc  = (const float*)d_in[9];
  const float* Wxo = (const float*)d_in[10];
  const float* Who = (const float*)d_in[11];
  const float* bo  = (const float*)d_in[12];

  // workspace layout (bytes); total ~304.2 MB
  char* ws = (char*)d_ws;
  float*          P    = (float*)ws;                              // 256 MB pre-activations
  unsigned short* WxBF = (unsigned short*)(ws + 268435456ull);    // 8 MB
  unsigned short* WhBF = (unsigned short*)(ws + 276824064ull);    // 8 MB
  unsigned short* XBF  = (unsigned short*)(ws + 285212672ull);    // 32 MB
  float*          BIAS = (float*)(ws + 318767104ull);             // 16 KB
  unsigned short* HBUF = (unsigned short*)(ws + 318783488ull);    // 128 KB
  unsigned int*   BAR  = (unsigned int*)(ws + 318914560ull);      // 128 B

  cast_x_kernel<<<(B_ * T_ * I_) / 256, 256, 0, stream>>>(x, XBF, B_ * T_ * I_);
  WPtrs wp = {{Wxi, Wxf, Wxc, Wxo, Whi, Whf, Whc, Who}};
  pack_w_kernel<<<(8 * H_ * I_) / 256, 256, 0, stream>>>(wp, WxBF, WhBF);
  BPtrs bp = {{bi, bf_, bc, bo}};
  pack_b_kernel<<<16, 256, 0, stream>>>(bp, BIAS);
  init_kernel<<<1, 256, 0, stream>>>(HBUF, BAR);

  xproj_kernel<<<8192, 256, 128 * 1024, stream>>>(XBF, WxBF, BIAS, P);
  lstm_scan_kernel<<<NWG_, 256, 136 * 1024, stream>>>(P, WhBF, HBUF, (float*)d_out, BAR);
}
